// GATDiscriminator_10024453668907
// MI455X (gfx1250) — matrix operand order repaired and checked by reading, not compiled, not run on hardware
//
#include <hip/hip_runtime.h>
#include <hip/hip_bf16.h>
#include <math.h>

typedef _Float16 v16h __attribute__((ext_vector_type(16)));
typedef _Float16 v8h  __attribute__((ext_vector_type(8)));
typedef float    v8f  __attribute__((ext_vector_type(8)));

#define FDIM 128
#define HEADS 2
#define HIDC 64
#define NEG_SLOPE 0.2f
#define EPSV 1e-16f
#define EPW 4   // edges per wave in scatter

__global__ void zero_f32(float* __restrict__ p, int n) {
    int i = blockIdx.x * blockDim.x + threadIdx.x;
    if (i < n) p[i] = 0.f;
}

// 8 elements per thread: two float4 loads -> one 16B v8h store
__global__ void f32_to_f16v(const float* __restrict__ in, _Float16* __restrict__ out, int n8) {
    int i = blockIdx.x * blockDim.x + threadIdx.x;
    if (i >= n8) return;
    const float4* p = (const float4*)in + (size_t)i * 2;
    float4 a = p[0], b = p[1];
    v8h o = { (_Float16)a.x, (_Float16)a.y, (_Float16)a.z, (_Float16)a.w,
              (_Float16)b.x, (_Float16)b.y, (_Float16)b.z, (_Float16)b.w };
    ((v8h*)out)[i] = o;
}

// W: [128 x 128] row-major (k, c) f32  ->  Wt: [c][k] f16 (column-major for WMMA B frags)
__global__ void w_to_f16_t(const float* __restrict__ W, _Float16* __restrict__ Wt) {
    int i = blockIdx.x * blockDim.x + threadIdx.x;
    if (i >= FDIM * FDIM) return;
    int k = i >> 7, c = i & (FDIM - 1);
    Wt[c * FDIM + k] = (_Float16)W[k * FDIM + c];
}

// C[M x 128] = A[M x 128] @ W[128 x 128]; A f16 row-major, Wt f16 col-major.
// One wave per 16x16 output tile; 8 waves/block cover all 128 columns of 16 rows.
// A slab (16 x 128 f16 = 4KB) staged in LDS once per block (shared by all 8 waves).
__global__ void gemm_wmma_f16(const _Float16* __restrict__ Ah,
                              const _Float16* __restrict__ Wt,
                              float* __restrict__ C, int M) {
    __shared__ _Float16 As[16 * FDIM];
    int tid  = threadIdx.x;
    int row0 = blockIdx.x * 16;

    // cooperative stage: 2048 halfs = 256 x b128, one per thread
    {
        const v8h* gsrc = (const v8h*)(Ah + (size_t)row0 * FDIM);
        ((v8h*)As)[tid] = gsrc[tid];
    }
    __syncthreads();

    int lane = tid & 31;
    int wid  = tid >> 5;                  // 0..7 -> column tile
    int col0 = wid * 16;
    int mrow = lane & 15;                 // M (for A) / N (for B) within tile
    int kkA  = (lane >> 4) * 16;          // A: low lanes K0-15, high lanes K16-31
    int kkB  = (lane >> 4) * 16;          // B: low lanes K0-15, high lanes K16-31

    const _Float16* arow = As + mrow * FDIM;
    const _Float16* bcol = Wt + (size_t)(col0 + mrow) * FDIM;

    v8f acc = {};
#pragma unroll
    for (int k = 0; k < FDIM; k += 32) {
        v8h alo = *(const v8h*)(arow + k + kkA);
        v8h ahi = *(const v8h*)(arow + k + kkA + 8);
        v16h a  = __builtin_shufflevector(alo, ahi,
                    0,1,2,3,4,5,6,7,8,9,10,11,12,13,14,15);
        v16h b  = *(const v16h*)(bcol + k + kkB);
        acc = __builtin_amdgcn_wmma_f32_16x16x32_f16(
                  false, a, false, b, (short)0, acc, false, false);
    }
    // C/D layout: VGPR v -> row v (+8 for high lanes), col = lane&15
    int cr = row0 + (lane >> 4) * 8;
    int cc = col0 + mrow;
#pragma unroll
    for (int v = 0; v < 8; ++v)
        C[(size_t)(cr + v) * FDIM + cc] = acc[v];
}

// a_src[n,h] = <h[n,h,:], att_src[h,:]>, same for a_dst. One thread per (node, head).
__global__ void attn_scores(const float* __restrict__ h,
                            const float* __restrict__ att_src,
                            const float* __restrict__ att_dst,
                            float* __restrict__ asrc, float* __restrict__ adst, int n) {
    int i = blockIdx.x * blockDim.x + threadIdx.x;
    if (i >= n * HEADS) return;
    int node = i >> 1, hd = i & 1;
    const float* hp = h + (size_t)node * FDIM + hd * HIDC;
    const float* ws = att_src + hd * HIDC;
    const float* wd = att_dst + hd * HIDC;
    float s = 0.f, d = 0.f;
#pragma unroll 8
    for (int c = 0; c < HIDC; ++c) { s += hp[c] * ws[c]; d += hp[c] * wd[c]; }
    asrc[i] = s;
    adst[i] = d;
}

// Pass 1: ssum[dst,h] += exp(leaky_relu(a_src[src,h] + a_dst[dst,h]))
// Edges >= E are the N appended self-loops.
__global__ void edge_expsum(const long long* __restrict__ src,
                            const long long* __restrict__ dst,
                            const float* __restrict__ asrc,
                            const float* __restrict__ adst,
                            float* __restrict__ ssum,
                            long long E, long long E2) {
    long long t = (long long)blockIdx.x * blockDim.x + threadIdx.x;
    if (t >= E2 * HEADS) return;
    long long e = t >> 1;
    int hd = (int)(t & 1);
    int s, d;
    if (e < E) { s = (int)src[e]; d = (int)dst[e]; }
    else       { s = d = (int)(e - E); }
    float v = asrc[s * 2 + hd] + adst[d * 2 + hd];
    v = v > 0.f ? v : NEG_SLOPE * v;
    atomicAdd(&ssum[d * 2 + hd], __expf(v));
}

// Pass 2: agg[dst,:] += alpha * h[src,:]; EPW edges per wave, 4 floats per lane.
// Prefetch the next edge's 512B feature row while current edge's atomics drain.
__global__ void edge_scatter(const long long* __restrict__ src,
                             const long long* __restrict__ dst,
                             const float* __restrict__ asrc,
                             const float* __restrict__ adst,
                             const float* __restrict__ ssum,
                             const float* __restrict__ h,
                             float* __restrict__ agg,
                             long long E, long long E2) {
    int lane = threadIdx.x & 31;
    long long wbase =
        ((long long)blockIdx.x * (blockDim.x >> 5) + (threadIdx.x >> 5)) * EPW;
#pragma unroll
    for (int i = 0; i < EPW; ++i) {
        long long wv = wbase + i;
        if (wv >= E2) return;
        // prefetch the next edge's source row (h is L2-resident; warm near caches)
        if (i + 1 < EPW && wv + 1 < E2) {
            long long nx = wv + 1;
            int sn = (nx < E) ? (int)src[nx] : (int)(nx - E);
            __builtin_prefetch(h + (size_t)sn * FDIM + lane * 4, 0, 1);
        }
        int s, d;
        if (wv < E) { s = (int)src[wv]; d = (int)dst[wv]; }
        else        { s = d = (int)(wv - E); }
        int hd = lane >> 4;  // features lane*4..lane*4+3 belong to head lane/16
        float v = asrc[s * 2 + hd] + adst[d * 2 + hd];
        v = v > 0.f ? v : NEG_SLOPE * v;
        float alpha = __expf(v) / (ssum[d * 2 + hd] + EPSV);
        const float4 hv = *(const float4*)(h + (size_t)s * FDIM + lane * 4);
        float* op = agg + (size_t)d * FDIM + lane * 4;
        atomicAdd(op + 0, hv.x * alpha);
        atomicAdd(op + 1, hv.y * alpha);
        atomicAdd(op + 2, hv.z * alpha);
        atomicAdd(op + 3, hv.w * alpha);
    }
}

__global__ void bias_relu(const float* __restrict__ agg, const float* __restrict__ b,
                          float* __restrict__ out, int total) {
    int i = blockIdx.x * blockDim.x + threadIdx.x;
    if (i >= total) return;
    float v = agg[i] + b[i & (FDIM - 1)];
    out[i] = v > 0.f ? v : 0.f;
}

// out[n] = sigmoid(<h[n,:], fcW> + fcb); one wave per node, shuffle reduction.
__global__ void fc_sigmoid(const float* __restrict__ h, const float* __restrict__ fcW,
                           const float* __restrict__ fcb, float* __restrict__ out, int n) {
    int lane = threadIdx.x & 31;
    int node = blockIdx.x * (blockDim.x >> 5) + (threadIdx.x >> 5);
    if (node >= n) return;
    const float4 hv = *(const float4*)(h + (size_t)node * FDIM + lane * 4);
    const float4 wv = *(const float4*)(fcW + lane * 4);
    float p = hv.x * wv.x + hv.y * wv.y + hv.z * wv.z + hv.w * wv.w;
#pragma unroll
    for (int off = 16; off > 0; off >>= 1) p += __shfl_down(p, off, 32);
    if (lane == 0) out[node] = 1.f / (1.f + __expf(-(p + fcb[0])));
}

extern "C" void kernel_launch(void* const* d_in, const int* in_sizes, int n_in,
                              void* d_out, int out_size, void* d_ws, size_t ws_size,
                              hipStream_t stream) {
    const float*     x   = (const float*)d_in[0];
    const long long* ei  = (const long long*)d_in[1];
    // d_in[2] = edge_weight: unused by GATConv math
    const float* W1  = (const float*)d_in[3];
    const float* as1 = (const float*)d_in[4];
    const float* ad1 = (const float*)d_in[5];
    const float* b1  = (const float*)d_in[6];
    const float* W2  = (const float*)d_in[7];
    const float* as2 = (const float*)d_in[8];
    const float* ad2 = (const float*)d_in[9];
    const float* b2  = (const float*)d_in[10];
    const float* fcW = (const float*)d_in[11];
    const float* fcb = (const float*)d_in[12];

    const int       N  = in_sizes[0] / FDIM;
    const long long E  = (long long)(in_sizes[1] / 2);
    const long long E2 = E + N;
    const long long* src = ei;
    const long long* dst = ei + E;

    const size_t nf = (size_t)N * FDIM;
    char* ws = (char*)d_ws;
    float*     B0   = (float*)ws;     ws += nf * sizeof(float);
    float*     B1   = (float*)ws;     ws += nf * sizeof(float);
    float*     B2   = (float*)ws;     ws += nf * sizeof(float);
    _Float16*  Ah   = (_Float16*)ws;  ws += nf * sizeof(_Float16);
    _Float16*  Wt   = (_Float16*)ws;  ws += (size_t)FDIM * FDIM * sizeof(_Float16);
    float*     asrc = (float*)ws;     ws += (size_t)N * HEADS * sizeof(float);
    float*     adst = (float*)ws;     ws += (size_t)N * HEADS * sizeof(float);
    float*     ssum = (float*)ws;     ws += (size_t)N * HEADS * sizeof(float);

    const dim3 blk(256);
    auto cdiv = [](long long a, long long b) { return (unsigned)((a + b - 1) / b); };

    auto layer = [&](const float* inF, const float* W, const float* as_,
                     const float* ad_, const float* b_, float* outF) {
        f32_to_f16v<<<cdiv((long long)nf / 8, 256), blk, 0, stream>>>(inF, Ah, (int)(nf / 8));
        w_to_f16_t <<<cdiv(FDIM * FDIM, 256),   blk, 0, stream>>>(W, Wt);
        gemm_wmma_f16<<<N / 16,                 blk, 0, stream>>>(Ah, Wt, B0, N);
        attn_scores<<<cdiv((long long)N * HEADS, 256), blk, 0, stream>>>(B0, as_, ad_, asrc, adst, N);
        zero_f32   <<<cdiv((long long)N * HEADS, 256), blk, 0, stream>>>(ssum, N * HEADS);
        zero_f32   <<<cdiv((long long)nf, 256), blk, 0, stream>>>(B1, (int)nf);
        edge_expsum<<<cdiv(E2 * HEADS, 256),    blk, 0, stream>>>(src, dst, asrc, adst, ssum, E, E2);
        edge_scatter<<<cdiv(E2, 8LL * EPW),     blk, 0, stream>>>(src, dst, asrc, adst, ssum, B0, B1, E, E2);
        bias_relu  <<<cdiv((long long)nf, 256), blk, 0, stream>>>(B1, b_, outF, (int)nf);
    };

    layer(x,  W1, as1, ad1, b1, B2);   // layer 1: x -> B2
    layer(B2, W2, as2, ad2, b2, B2);   // layer 2: B2 -> B2 (safe: input copied to f16 first)
    fc_sigmoid<<<cdiv(N, 8), blk, 0, stream>>>(B2, fcW, fcb, (float*)d_out, N);
}
